// MultiHeadAttention_12738873000258
// MI455X (gfx1250) — matrix operand order repaired and checked by reading, not compiled, not run on hardware
//
#include <hip/hip_runtime.h>
#include <hip/hip_bf16.h>

typedef __attribute__((ext_vector_type(16))) _Float16 v16h;
typedef __attribute__((ext_vector_type(8)))  _Float16 v8h;
typedef __attribute__((ext_vector_type(4)))  _Float16 v4h;
typedef __attribute__((ext_vector_type(8)))  float    v8f;

#define SHIFT 4

union FragU { v16h v; v8h h[2]; };

__device__ __forceinline__ v16h frag2(const _Float16* p0, const _Float16* p1) {
  FragU u; u.h[0] = *(const v8h*)p0; u.h[1] = *(const v8h*)p1; return u.v;
}
__device__ __forceinline__ v8f zero8() {
  v8f z; for (int i = 0; i < 8; ++i) z[i] = 0.0f; return z;
}

// ------------------------------------------------------------------
// 1) weight f32 -> f16 convert.  W16 = [Wq;Wk;Wv] rows 0..767, k=0..255
// ------------------------------------------------------------------
__global__ __launch_bounds__(256) void k_convw(
    const float* __restrict__ Wq, const float* __restrict__ Wk,
    const float* __restrict__ Wv, const float* __restrict__ Wo,
    _Float16* __restrict__ W16, _Float16* __restrict__ Wo16) {
  int i = blockIdx.x * 256 + threadIdx.x;
  if (i < 65536) {
    W16[i]          = (_Float16)Wq[i];
    W16[65536 + i]  = (_Float16)Wk[i];
    W16[131072 + i] = (_Float16)Wv[i];
    Wo16[i]         = (_Float16)Wo[i];
  }
}

// ------------------------------------------------------------------
// 2) CPB-MLP bias table: biasTab[h][m][n]  (m = query token, n = key)
// ------------------------------------------------------------------
__device__ __forceinline__ float cpbcoord(int d) {  // d in [-7,7]
  float t = (float)d * (8.0f / 7.0f);
  float s = (t > 0.f) ? 1.f : ((t < 0.f) ? -1.f : 0.f);
  return s * log2f(fabsf(t) + 1.f) * (1.0f / 3.0f);  // /log2(8)
}

__global__ __launch_bounds__(256) void k_cpb(
    const float* __restrict__ w1, const float* __restrict__ b1,
    const float* __restrict__ w2, float* __restrict__ biasTab) {
  __shared__ float outt[225][8];
  int tid = threadIdx.x;
  if (tid < 225) {
    int i = tid / 15, j = tid % 15;
    float t0 = cpbcoord(i - 7), t1 = cpbcoord(j - 7);
    float acc[8];
    for (int h = 0; h < 8; ++h) acc[h] = 0.f;
    for (int d = 0; d < 512; ++d) {
      float hv = fmaxf(0.f, w1[d * 2] * t0 + w1[d * 2 + 1] * t1 + b1[d]);
      for (int h = 0; h < 8; ++h) acc[h] += w2[h * 512 + d] * hv;
    }
    for (int h = 0; h < 8; ++h) outt[tid][h] = 16.f / (1.f + __expf(-acc[h]));
  }
  __syncthreads();
  for (int e = tid; e < 8 * 64 * 64; e += 256) {
    int h = e >> 12, m = (e >> 6) & 63, n = e & 63;
    int dy = (m >> 3) - (n >> 3) + 7;
    int dx = (m & 7) - (n & 7) + 7;
    biasTab[e] = outt[dy * 15 + dx][h];
  }
}

// ------------------------------------------------------------------
// 3) QKV projection per window. Gathers shifted x, GEMM 64x768x256.
//    Q,K -> [win][64][256] f16 ; V -> transposed [win][256][64] f16
//    m-tiles processed in pairs (bounds live A-fragments),
//    B fragments double-buffered to overlap global loads with WMMA.
// ------------------------------------------------------------------
__global__ __launch_bounds__(256) void k_qkv(
    const float* __restrict__ x, const _Float16* __restrict__ W16,
    const float* __restrict__ bq, const float* __restrict__ bk,
    const float* __restrict__ bv,
    _Float16* __restrict__ Q, _Float16* __restrict__ K,
    _Float16* __restrict__ VT) {
  __shared__ __align__(16) _Float16 lx[64][264];
  int wi = blockIdx.x;
  int b = wi >> 6, win = wi & 63, wy = win >> 3, wx = win & 7;
  {
    int row = threadIdx.x >> 2, part = threadIdx.x & 3;
    int y  = (wy * 8 + (row >> 3) + SHIFT) & 63;
    int xx = (wx * 8 + (row & 7) + SHIFT) & 63;
    const float4* src =
        (const float4*)(x + (((size_t)(b * 64 + y)) * 64 + xx) * 256) + part * 16;
    for (int j = 0; j < 16; ++j) {
      float4 f = src[j];
      int c = part * 64 + j * 4;
      v4h p;
      p[0] = (_Float16)f.x; p[1] = (_Float16)f.y;
      p[2] = (_Float16)f.z; p[3] = (_Float16)f.w;
      *(v4h*)&lx[row][c] = p;
    }
  }
  __syncthreads();
  int wave = threadIdx.x >> 5, lane = threadIdx.x & 31;
  int lrow = lane & 15, lhi = lane >> 4;
#pragma clang loop unroll(disable)
  for (int mh = 0; mh < 2; ++mh) {        // m-tile pair {0,1} / {2,3}
    v16h Af[2][8];
    for (int mt2 = 0; mt2 < 2; ++mt2)
      for (int ks = 0; ks < 8; ++ks) {
        const _Float16* ap = &lx[(mh * 2 + mt2) * 16 + lrow][ks * 32 + lhi * 8];
        Af[mt2][ks] = frag2(ap, ap + 16);
      }
    for (int i = 0; i < 6; ++i) {
      int nt = wave + i * 8;          // 0..47
      int ngl = nt * 16 + lrow;       // output feature 0..767
      const _Float16* wbase = W16 + (size_t)ngl * 256 + lhi * 8;
      v8f acc0 = zero8(), acc1 = zero8();
      v16h Bf = frag2(wbase, wbase + 16);
      for (int ks = 0; ks < 8; ++ks) {
        v16h Bn;
        if (ks < 7) {                  // prefetch next B while WMMAs run
          const _Float16* wp = wbase + (ks + 1) * 32;
          Bn = frag2(wp, wp + 16);
        }
        acc0 = __builtin_amdgcn_wmma_f32_16x16x32_f16(
            false, Af[0][ks], false, Bf, (short)0, acc0, false, false);
        acc1 = __builtin_amdgcn_wmma_f32_16x16x32_f16(
            false, Af[1][ks], false, Bf, (short)0, acc1, false, false);
        if (ks < 7) Bf = Bn;
      }
      int mat = ngl >> 8;             // 0=q 1=k 2=v
      int nc = ngl & 255;
      float bb = (mat == 0) ? bq[nc] : (mat == 1) ? bk[nc] : bv[nc];
      int m0 = (mh * 2) * 16 + lhi * 8;
      int m1 = (mh * 2 + 1) * 16 + lhi * 8;
      if (mat < 2) {
        _Float16* dst = ((mat == 0) ? Q : K) + (size_t)wi * 16384 + nc;
        for (int r = 0; r < 8; ++r) {
          dst[(m0 + r) * 256] = (_Float16)(acc0[r] + bb);
          dst[(m1 + r) * 256] = (_Float16)(acc1[r] + bb);
        }
      } else {
        _Float16* dst = VT + (size_t)wi * 16384 + (size_t)nc * 64;
        v8h pk0, pk1;
        for (int r = 0; r < 8; ++r) {
          pk0[r] = (_Float16)(acc0[r] + bb);
          pk1[r] = (_Float16)(acc1[r] + bb);
        }
        *(v8h*)(dst + m0) = pk0;
        *(v8h*)(dst + m1) = pk1;
      }
    }
  }
}

// ------------------------------------------------------------------
// 4) Attention: one wave per (window, head).
// ------------------------------------------------------------------
__global__ __launch_bounds__(256) void k_attn(
    const _Float16* __restrict__ Q, const _Float16* __restrict__ K,
    const _Float16* __restrict__ VT, const float* __restrict__ biasTab,
    const float* __restrict__ logit_scale, _Float16* __restrict__ AO) {
  __shared__ __align__(16) _Float16 lp[8][16][72];
  __shared__ float qinv[8][64];
  __shared__ float kinv[8][64];
  __shared__ int   regs[64];
  int wi = blockIdx.x;
  int win = wi & 63, wy = win >> 3, wx = win & 7;
  int head = threadIdx.x >> 5, lane = threadIdx.x & 31;
  int lrow = lane & 15, lhi = lane >> 4;
  const _Float16* qh = Q + (size_t)wi * 16384 + head * 32;
  const _Float16* kh = K + (size_t)wi * 16384 + head * 32;
  if (threadIdx.x < 64) {
    int t = threadIdx.x;
    int ys = wy * 8 + (t >> 3), xs = wx * 8 + (t & 7);
    int ry = (ys < 56) ? 0 : ((ys < 60) ? 1 : 2);
    int rx = (xs < 56) ? 0 : ((xs < 60) ? 1 : 2);
    regs[t] = ry * 3 + rx;
  }
  for (int rr = 0; rr < 2; ++rr) {
    int row = lane * 2 + rr;
    const v8h* qp = (const v8h*)(qh + row * 256);
    const v8h* kp = (const v8h*)(kh + row * 256);
    float sq = 0.f, sk = 0.f;
    for (int j = 0; j < 4; ++j) {
      v8h a = qp[j], c = kp[j];
      for (int e = 0; e < 8; ++e) {
        float fa = (float)a[e]; sq += fa * fa;
        float fb = (float)c[e]; sk += fb * fb;
      }
    }
    qinv[head][row] = rsqrtf(fmaxf(sq, 1e-24f));
    kinv[head][row] = rsqrtf(fmaxf(sk, 1e-24f));
  }
  __syncthreads();
  float lsc = __expf(fminf(logit_scale[head], 4.6051702f));  // log(100)
  const _Float16* vt = VT + (size_t)wi * 16384 + (size_t)head * 32 * 64;
  for (int mt = 0; mt < 4; ++mt) {
    v8f S[4];
    for (int nt = 0; nt < 4; ++nt) S[nt] = zero8();
    {
      const _Float16* ap = qh + (mt * 16 + lrow) * 256 + lhi * 8;
      v16h Af = frag2(ap, ap + 16);
      for (int nt = 0; nt < 4; ++nt) {
        const _Float16* bp = kh + (nt * 16 + lrow) * 256 + lhi * 8;
        v16h Bf = frag2(bp, bp + 16);
        S[nt] = __builtin_amdgcn_wmma_f32_16x16x32_f16(
            false, Af, false, Bf, (short)0, S[nt], false, false);
      }
    }
    int regn[4]; float kv[4];
    for (int nt = 0; nt < 4; ++nt) {
      regn[nt] = regs[nt * 16 + lrow];
      kv[nt] = kinv[head][nt * 16 + lrow];
    }
    for (int r = 0; r < 8; ++r) {
      int ml = r + 8 * lhi;
      int m = mt * 16 + ml;
      float qi = qinv[head][m] * lsc;
      int rm = regs[m];
      const float* bt = biasTab + head * 4096 + m * 64 + lrow;
      float vals[4];
      float mx = -1e30f;
      for (int nt = 0; nt < 4; ++nt) {
        float v = S[nt][r] * qi * kv[nt] + bt[nt * 16];
        if (rm != regn[nt]) v -= 100.f;
        vals[nt] = v;
        mx = fmaxf(mx, v);
      }
      for (int o = 1; o < 16; o <<= 1) mx = fmaxf(mx, __shfl_xor(mx, o, 32));
      float sum = 0.f;
      for (int nt = 0; nt < 4; ++nt) { vals[nt] = __expf(vals[nt] - mx); sum += vals[nt]; }
      for (int o = 1; o < 16; o <<= 1) sum += __shfl_xor(sum, o, 32);
      float inv = 1.0f / sum;
      for (int nt = 0; nt < 4; ++nt)
        lp[head][ml][nt * 16 + lrow] = (_Float16)(vals[nt] * inv);
    }
    __syncthreads();
    v8f O[2]; O[0] = zero8(); O[1] = zero8();
    for (int ks = 0; ks < 2; ++ks) {
      const _Float16* ap = &lp[head][lrow][ks * 32 + lhi * 8];
      v16h Af = frag2(ap, ap + 16);
      for (int nt = 0; nt < 2; ++nt) {
        const _Float16* bp = vt + (nt * 16 + lrow) * 64 + ks * 32 + lhi * 8;
        v16h Bf = frag2(bp, bp + 16);
        O[nt] = __builtin_amdgcn_wmma_f32_16x16x32_f16(
            false, Af, false, Bf, (short)0, O[nt], false, false);
      }
    }
    _Float16* dst = AO + (size_t)wi * 16384 + head * 32;
    for (int nt = 0; nt < 2; ++nt)
      for (int r = 0; r < 8; ++r) {
        int m = mt * 16 + r + 8 * lhi;
        dst[m * 256 + nt * 16 + lrow] = (_Float16)O[nt][r];
      }
    __syncthreads();
  }
}

// ------------------------------------------------------------------
// 5) Output projection + window reverse + reverse shift -> f32 out
// ------------------------------------------------------------------
__global__ __launch_bounds__(256) void k_oproj(
    const _Float16* __restrict__ AO, const _Float16* __restrict__ Wo16,
    const float* __restrict__ bo, float* __restrict__ out) {
  __shared__ __align__(16) _Float16 la[64][264];
  int wi = blockIdx.x;
  int b = wi >> 6, win = wi & 63, wy = win >> 3, wx = win & 7;
  {
    int row = threadIdx.x >> 2, part = threadIdx.x & 3;
    const v8h* src = (const v8h*)(AO + (size_t)wi * 16384 + row * 256 + part * 64);
    v8h* dst = (v8h*)&la[row][part * 64];
    for (int j = 0; j < 8; ++j) dst[j] = src[j];
  }
  __syncthreads();
  int wave = threadIdx.x >> 5, lane = threadIdx.x & 31;
  int lrow = lane & 15, lhi = lane >> 4;
#pragma clang loop unroll(disable)
  for (int mh = 0; mh < 2; ++mh) {        // m-tile pair
    v16h Af[2][8];
    for (int mt2 = 0; mt2 < 2; ++mt2)
      for (int ks = 0; ks < 8; ++ks) {
        const _Float16* ap = &la[(mh * 2 + mt2) * 16 + lrow][ks * 32 + lhi * 8];
        Af[mt2][ks] = frag2(ap, ap + 16);
      }
    for (int i = 0; i < 2; ++i) {
      int nt = wave * 2 + i;
      int n = nt * 16 + lrow;
      const _Float16* wbase = Wo16 + (size_t)n * 256 + lhi * 8;
      v8f acc0 = zero8(), acc1 = zero8();
      v16h Bf = frag2(wbase, wbase + 16);
      for (int ks = 0; ks < 8; ++ks) {
        v16h Bn;
        if (ks < 7) {
          const _Float16* wp = wbase + (ks + 1) * 32;
          Bn = frag2(wp, wp + 16);
        }
        acc0 = __builtin_amdgcn_wmma_f32_16x16x32_f16(
            false, Af[0][ks], false, Bf, (short)0, acc0, false, false);
        acc1 = __builtin_amdgcn_wmma_f32_16x16x32_f16(
            false, Af[1][ks], false, Bf, (short)0, acc1, false, false);
        if (ks < 7) Bf = Bn;
      }
      float bb = bo[n];
      for (int mt2 = 0; mt2 < 2; ++mt2) {
        for (int r = 0; r < 8; ++r) {
          int m = (mh * 2 + mt2) * 16 + lhi * 8 + r;
          int y  = (wy * 8 + (m >> 3) + SHIFT) & 63;
          int xx = (wx * 8 + (m & 7) + SHIFT) & 63;
          float v = (mt2 == 0 ? acc0[r] : acc1[r]) + bb;
          out[(((size_t)(b * 64 + y)) * 64 + xx) * 256 + n] = v;
        }
      }
    }
  }
}

// ------------------------------------------------------------------
extern "C" void kernel_launch(void* const* d_in, const int* in_sizes, int n_in,
                              void* d_out, int out_size, void* d_ws, size_t ws_size,
                              hipStream_t stream) {
  const float* x  = (const float*)d_in[0];
  const float* Wq = (const float*)d_in[1];
  const float* bq = (const float*)d_in[2];
  const float* Wk = (const float*)d_in[3];
  const float* bk = (const float*)d_in[4];
  const float* Wv = (const float*)d_in[5];
  const float* bv = (const float*)d_in[6];
  const float* Wo = (const float*)d_in[7];
  const float* bo = (const float*)d_in[8];
  const float* ls = (const float*)d_in[9];
  const float* w1 = (const float*)d_in[10];
  const float* b1 = (const float*)d_in[11];
  const float* w2 = (const float*)d_in[12];
  float* out = (float*)d_out;

  char* ws = (char*)d_ws;
  size_t off = 0;
  auto take = [&](size_t bytes) -> char* {
    char* p = ws + off;
    off += (bytes + 255) & ~(size_t)255;
    return p;
  };
  _Float16* W16     = (_Float16*)take((size_t)768 * 256 * sizeof(_Float16));
  _Float16* Wo16    = (_Float16*)take((size_t)256 * 256 * sizeof(_Float16));
  float*    biasTab = (float*)take((size_t)8 * 64 * 64 * sizeof(float));
  const size_t WBUF = (size_t)2048 * 16384 * sizeof(_Float16);  // 64 MiB
  _Float16* Qb = (_Float16*)take(WBUF);
  _Float16* Kb = (_Float16*)take(WBUF);
  _Float16* VT = (_Float16*)take(WBUF);
  _Float16* AO = (_Float16*)take(WBUF);

  k_convw<<<256, 256, 0, stream>>>(Wq, Wk, Wv, Wo, W16, Wo16);
  k_cpb<<<1, 256, 0, stream>>>(w1, b1, w2, biasTab);
  k_qkv<<<2048, 256, 0, stream>>>(x, W16, bq, bk, bv, Qb, Kb, VT);
  k_attn<<<2048, 256, 0, stream>>>(Qb, Kb, VT, biasTab, ls, AO);
  k_oproj<<<2048, 256, 0, stream>>>(AO, Wo16, bo, out);
}
